// LSTNet_AttSkip_10685878632621
// MI455X (gfx1250) — matrix operand order repaired and checked by reading, not compiled, not run on hardware
//
#include <hip/hip_runtime.h>
#include <cmath>

// Sizes from the reference
#define BB 256
#define WW 168
#define DD 321
#define KK 6
#define HC 256
#define HG 256
#define HS 64
#define TT 24
#define ARN 24
#define SWN 8
#define KD 1926    // KK*DD
#define WL 167     // WW-1
#define XSTR 352   // padded d-stride (ushorts), multiple of 32
#define WPAD 5     // zero time-rows at front of xb (causal pad)
#define GPAD 24    // zero time-slices at front of gbufb (skip-GRU pad)

typedef unsigned short ushort_t;
typedef __attribute__((ext_vector_type(16))) __bf16         v16bf;
typedef __attribute__((ext_vector_type(8)))  unsigned int   v8u;
typedef __attribute__((ext_vector_type(4)))  unsigned int   v4u;
typedef __attribute__((ext_vector_type(8)))  float          v8f;

// Pack two f32 into one dword of two bf16 (truncation) with a single v_perm_b32.
__device__ __forceinline__ unsigned pk2(float lo, float hi) {
  return __builtin_amdgcn_perm(__builtin_bit_cast(unsigned, hi),
                               __builtin_bit_cast(unsigned, lo), 0x07060302u);
}
__device__ __forceinline__ ushort_t f2bfu(float f) {
  return (ushort_t)(__builtin_bit_cast(unsigned, f) >> 16);
}
__device__ __forceinline__ float bf2f(ushort_t u) {
  return __builtin_bit_cast(float, (unsigned)u << 16);
}

__device__ __forceinline__ v8f wmma_bf16(v16bf a, v16bf b, v8f c) {
  return __builtin_amdgcn_wmma_f32_16x16x32_bf16(false, a, false, b, (short)0, c,
                                                 false, false);
}

// A-fragment: elems 0..15 = p[0..15]  (bf16, 16B aligned)
__device__ __forceinline__ v16bf fragA(const ushort_t* p) {
  v4u a = *(const v4u*)p;
  v4u b = *(const v4u*)(p + 8);
  v8u r; r[0]=a[0]; r[1]=a[1]; r[2]=a[2]; r[3]=a[3];
         r[4]=b[0]; r[5]=b[1]; r[6]=b[2]; r[7]=b[3];
  return __builtin_bit_cast(v16bf, r);
}
// B-fragment: 16 contiguous bf16 (32B, 16B aligned)
__device__ __forceinline__ v16bf fragB(const ushort_t* p) {
  v4u a = *(const v4u*)p;
  v4u b = *(const v4u*)(p + 8);
  v8u r; r[0]=a[0]; r[1]=a[1]; r[2]=a[2]; r[3]=a[3];
         r[4]=b[0]; r[5]=b[1]; r[6]=b[2]; r[7]=b[3];
  return __builtin_bit_cast(v16bf, r);
}

__device__ __forceinline__ float sigm(float x) { return 1.0f / (1.0f + __expf(-x)); }

__global__ void zero_f(float* p, int n) {
  int i = blockIdx.x * blockDim.x + threadIdx.x;
  if (i < n) p[i] = 0.0f;
}
__global__ void zero_u(unsigned* p, int n) {
  int i = blockIdx.x * blockDim.x + threadIdx.x;
  if (i < n) p[i] = 0u;
}

// Generic f32 -> packed bf16 (two per dword); ndw = element_count/2 (even counts only)
__global__ void pack_bf16(const float* __restrict__ src, unsigned* __restrict__ dst, int ndw) {
  int i = blockIdx.x * blockDim.x + threadIdx.x;
  if (i < ndw) dst[i] = pk2(src[2 * i], src[2 * i + 1]);
}

// x -> xb[b][wp][d]: wp<WPAD rows zero, d-stride XSTR with zero tail
__global__ void pack_x(const float* __restrict__ x, unsigned* __restrict__ xb) {
  int i = blockIdx.x * blockDim.x + threadIdx.x;
  if (i >= BB * (WW + WPAD) * (XSTR / 2)) return;
  int row = i / (XSTR / 2);
  int dd  = (i - row * (XSTR / 2)) * 2;
  int b   = row / (WW + WPAD);
  int wp  = row - b * (WW + WPAD);
  unsigned v = 0u;
  if (wp >= WPAD && dd < DD) {
    const float* xr = x + ((size_t)b * WW + (wp - WPAD)) * DD;
    float f0 = xr[dd];
    float f1 = (dd + 1 < DD) ? xr[dd + 1] : 0.0f;
    v = pk2(f0, f1);
  }
  xb[i] = v;
}

// conv_w -> cwb[n][kt][d], d-stride XSTR with zero tail
__global__ void pack_cw(const float* __restrict__ cw, unsigned* __restrict__ cwb) {
  int i = blockIdx.x * blockDim.x + threadIdx.x;
  if (i >= HC * KK * (XSTR / 2)) return;
  int row = i / (XSTR / 2);              // n*KK + kt
  int dd  = (i - row * (XSTR / 2)) * 2;
  int n   = row / KK, kt = row - n * KK;
  unsigned v = 0u;
  if (dd < DD) {
    const float* cr = cw + (size_t)n * KD + kt * DD;
    float f0 = cr[dd];
    float f1 = (dd + 1 < DD) ? cr[dd + 1] : 0.0f;
    v = pk2(f0, f1);
  }
  cwb[i] = v;
}

// ---------------------------------------------------------------------------
// Conv as 6 accumulated GEMMs over padded bf16 tensors; no masks, no tail.
// M = B*W = 43008 (32 rows/wave), N = 256 (64 cols/wave), K = 6*352.
// Stores relu(.+bias) as bf16 into gbufb at time-slice GPAD + w.
// ---------------------------------------------------------------------------
__global__ void conv_gemm(const ushort_t* __restrict__ xb, const ushort_t* __restrict__ cwb,
                          const float* __restrict__ cb, ushort_t* __restrict__ gbufb) {
  const int lane  = threadIdx.x & 31;
  const int wave  = (blockIdx.x * blockDim.x + threadIdx.x) >> 5;
  const int mpair = wave >> 2;            // 1344 groups of 32 rows
  const int n0    = (wave & 3) * 64;
  const int m0    = mpair * 32;
  const int ln    = lane & 15;
  const int hi    = lane >> 4;

  const ushort_t* axb[2];
#pragma unroll
  for (int u = 0; u < 2; ++u) {
    int mrow = m0 + u * 16 + ln;
    int bidx = mrow / WW;
    int wpos = mrow - bidx * WW;
    axb[u] = xb + ((size_t)bidx * (WW + WPAD) + wpos) * XSTR + hi * 16;
  }

  v8f zero8 = {0,0,0,0,0,0,0,0};
  v8f acc[2][4];
#pragma unroll
  for (int u = 0; u < 2; ++u)
#pragma unroll
    for (int t = 0; t < 4; ++t) acc[u][t] = zero8;

  for (int kt = 0; kt < KK; ++kt) {
    const ushort_t* a0 = axb[0] + kt * XSTR;     // row wp = wpos + kt
    const ushort_t* a1 = axb[1] + kt * XSTR;
    const ushort_t* bw = cwb + kt * XSTR + hi * 16;
    for (int d0 = 0; d0 < XSTR; d0 += 32) {
      __builtin_prefetch(bw + (size_t)(n0 + ln) * (KK * XSTR) + d0 + 256, 0, 1);
      v16bf af0 = fragA(a0 + d0);
      v16bf af1 = fragA(a1 + d0);
#pragma unroll
      for (int t = 0; t < 4; ++t) {
        int n = n0 + t * 16 + ln;
        v16bf bf = fragB(bw + (size_t)n * (KK * XSTR) + d0);
        acc[0][t] = wmma_bf16(af0, bf, acc[0][t]);
        acc[1][t] = wmma_bf16(af1, bf, acc[1][t]);
      }
    }
  }

#pragma unroll
  for (int u = 0; u < 2; ++u)
#pragma unroll
    for (int t = 0; t < 4; ++t) {
      int n = n0 + t * 16 + ln;
      float bias = cb[n];
#pragma unroll
      for (int r = 0; r < 8; ++r) {
        int m  = m0 + u * 16 + r + hi * 8;
        int b2 = m / WW, w2 = m - b2 * WW;
        float v = acc[u][t][r] + bias;
        v = v > 0.0f ? v : 0.0f;
        gbufb[((size_t)(GPAD + w2) * BB + b2) * HC + n] = f2bfu(v);
      }
    }
}

// ---------------------------------------------------------------------------
// LSTM gates: gates = [x_t | h] @ [Wih ; Whh]^T + bih + bhh  (all-bf16 operands)
// M=256, N=1024, K=512. 256 waves, 16x64 tiles.
// ---------------------------------------------------------------------------
__global__ void lstm_gates(const ushort_t* __restrict__ xt, const ushort_t* __restrict__ hb,
                           const ushort_t* __restrict__ wihb, const ushort_t* __restrict__ whhb,
                           const float* __restrict__ bih, const float* __restrict__ bhh,
                           float* __restrict__ gates) {
  const int lane  = threadIdx.x & 31;
  const int wave  = (blockIdx.x * blockDim.x + threadIdx.x) >> 5;
  const int mtile = wave >> 4;
  const int n0    = (wave & 15) * 64;
  const int m0    = mtile * 16;
  const int ln    = lane & 15;
  const int hi    = lane >> 4;
  const int arow  = m0 + ln;

  v8f zero8 = {0,0,0,0,0,0,0,0};
  v8f acc[4];
#pragma unroll
  for (int t = 0; t < 4; ++t) acc[t] = zero8;

  const ushort_t* ax = xt + (size_t)arow * HC + hi * 16;
  const ushort_t* ah = hb + (size_t)arow * HG + hi * 16;

  for (int k0 = 0; k0 < HC; k0 += 32) {
    v16bf af = fragA(ax + k0);
#pragma unroll
    for (int t = 0; t < 4; ++t) {
      int n = n0 + t * 16 + ln;
      acc[t] = wmma_bf16(af, fragB(wihb + (size_t)n * HC + k0 + hi * 16), acc[t]);
    }
  }
  for (int k0 = 0; k0 < HG; k0 += 32) {
    v16bf af = fragA(ah + k0);
#pragma unroll
    for (int t = 0; t < 4; ++t) {
      int n = n0 + t * 16 + ln;
      acc[t] = wmma_bf16(af, fragB(whhb + (size_t)n * HG + k0 + hi * 16), acc[t]);
    }
  }

#pragma unroll
  for (int t = 0; t < 4; ++t) {
    int n = n0 + t * 16 + ln;
    float bias = bih[n] + bhh[n];
#pragma unroll
    for (int r = 0; r < 8; ++r) {
      int m = m0 + r + hi * 8;
      gates[m * (4 * HG) + n] = acc[t][r] + bias;
    }
  }
}

// LSTM pointwise state update; writes h as bf16 (next-step GEMM operand) + f32 trace
__global__ void lstm_point(const float* __restrict__ gates, ushort_t* __restrict__ hb,
                           float* __restrict__ C, float* __restrict__ outs) {
  int idx = blockIdx.x * blockDim.x + threadIdx.x;   // 65536
  int b = idx >> 8, j = idx & 255;
  const float* gr = gates + b * (4 * HG);
  float gi = gr[j], gf = gr[HG + j], gg = gr[2 * HG + j], go = gr[3 * HG + j];
  float Cn = sigm(gf) * C[idx] + sigm(gi) * tanhf(gg);
  float hn = sigm(go) * tanhf(Cn);
  C[idx] = Cn;
  hb[idx] = f2bfu(hn);
  if (outs) outs[idx] = hn;
}

// attention scores: scores[b][t] = outs[t,b,:] . att_w + att_b
__global__ void att_score(const float* __restrict__ outs, const float* __restrict__ aw,
                          const float* __restrict__ ab, float* __restrict__ scores) {
  int idx = blockIdx.x * blockDim.x + threadIdx.x;   // WL*256
  int t = idx >> 8, b = idx & 255;
  const float4* o4 = (const float4*)(outs + ((size_t)t * BB + b) * HG);
  const float4* w4 = (const float4*)aw;
  float s = ab[0];
#pragma unroll 4
  for (int j = 0; j < HG / 4; ++j) {
    float4 o = o4[j], w = w4[j];
    s += o.x * w.x + o.y * w.y + o.z * w.z + o.w * w.w;
  }
  scores[b * WL + t] = s;
}

// softmax over t + weighted pool + last-step residual; writes bf16 att_out
__global__ void att_pool(const float* __restrict__ outs, const float* __restrict__ scores,
                         ushort_t* __restrict__ att_outb) {
  __shared__ float p[WL];
  __shared__ float inv_s;
  int b = blockIdx.x, j = threadIdx.x;
  if (j < WL) p[j] = scores[b * WL + j];
  __syncthreads();
  if (j == 0) {
    float m = p[0];
    for (int t = 1; t < WL; ++t) m = fmaxf(m, p[t]);
    float s = 0.0f;
    for (int t = 0; t < WL; ++t) { p[t] = __expf(p[t] - m); s += p[t]; }
    inv_s = 1.0f / s;
  }
  __syncthreads();
  float inv = inv_s;
  float acc = 0.0f;
  for (int t = 0; t < WL; ++t) acc += outs[((size_t)t * BB + b) * HG + j] * (p[t] * inv);
  acc += outs[((size_t)(WL - 1) * BB + b) * HG + j];
  att_outb[b * HG + j] = f2bfu(acc);
}

// res = h_f @ lg_w^T + lg_b.  M=256, N=321 (clamped), K=256. Writes d_out.
__global__ void lin_g(const ushort_t* __restrict__ hb, const ushort_t* __restrict__ wb,
                      const float* __restrict__ bias, float* __restrict__ out) {
  const int lane  = threadIdx.x & 31;
  const int wave  = (blockIdx.x * blockDim.x + threadIdx.x) >> 5;
  const int mtile = wave / 6;
  const int n0    = (wave % 6) * 64;
  const int m0    = mtile * 16;
  const int ln    = lane & 15;
  const int hi    = lane >> 4;
  const int arow  = m0 + ln;

  v8f zero8 = {0,0,0,0,0,0,0,0};
  v8f acc[4];
#pragma unroll
  for (int t = 0; t < 4; ++t) acc[t] = zero8;

  const ushort_t* ax = hb + (size_t)arow * HG + hi * 16;
  for (int k0 = 0; k0 < HG; k0 += 32) {
    v16bf af = fragA(ax + k0);
#pragma unroll
    for (int t = 0; t < 4; ++t) {
      int n  = n0 + t * 16 + ln;
      int nc = n < DD ? n : DD - 1;      // clamp: dead columns never stored
      acc[t] = wmma_bf16(af, fragB(wb + (size_t)nc * HG + k0 + hi * 16), acc[t]);
    }
  }
#pragma unroll
  for (int t = 0; t < 4; ++t) {
    int n = n0 + t * 16 + ln;
    if (n < DD) {
      float bias_n = bias[n];
#pragma unroll
      for (int r = 0; r < 8; ++r) {
        int m = m0 + r + hi * 8;
        out[m * DD + n] = acc[t][r] + bias_n;
      }
    }
  }
}

// GRU x-gates for all 8 skip steps; pad slices of gbufb supply the zero history.
// M=49152, N=192, K=256.
__global__ void gru_gx(const ushort_t* __restrict__ gbufb, const ushort_t* __restrict__ wihb,
                       const float* __restrict__ bih, float* __restrict__ gx) {
  const int lane  = threadIdx.x & 31;
  const int wave  = (blockIdx.x * blockDim.x + threadIdx.x) >> 5;
  const int mtile = wave / 3;
  const int n0    = (wave % 3) * 64;
  const int m0    = mtile * 16;
  const int ln    = lane & 15;
  const int hi    = lane >> 4;

  const int row = m0 + ln;
  const int sw  = row / (BB * TT);
  const int bt  = row - sw * (BB * TT);
  const int b   = bt / TT;
  const int t   = bt - b * TT;
  const int tp  = t * SWN + sw;           // 0..191; tp<GPAD are the zero pads
  const ushort_t* ax = gbufb + ((size_t)tp * BB + b) * HC + hi * 16;

  v8f zero8 = {0,0,0,0,0,0,0,0};
  v8f acc[4];
#pragma unroll
  for (int q = 0; q < 4; ++q) acc[q] = zero8;

  for (int k0 = 0; k0 < HC; k0 += 32) {
    v16bf af = fragA(ax + k0);
#pragma unroll
    for (int q = 0; q < 4; ++q) {
      int n = n0 + q * 16 + ln;
      acc[q] = wmma_bf16(af, fragB(wihb + (size_t)n * HC + k0 + hi * 16), acc[q]);
    }
  }
#pragma unroll
  for (int q = 0; q < 4; ++q) {
    int n = n0 + q * 16 + ln;
    float bias = bih[n];
#pragma unroll
    for (int r = 0; r < 8; ++r) {
      int mm = m0 + r + hi * 8;
      gx[mm * (3 * HS) + n] = acc[q][r] + bias;
    }
  }
}

// One fused GRU step: gh = h@Whh^T (+bhh) in registers (12 WMMA tiles), then r/z/n
// update in the owning lane. Each wave owns rows [m0, m0+16) exclusively.
__global__ void gru_step(ushort_t* __restrict__ hgb, const ushort_t* __restrict__ whhb,
                         const float* __restrict__ bhh, const float* __restrict__ gx,
                         int sw) {
  const int lane = threadIdx.x & 31;
  const int wave = (blockIdx.x * blockDim.x + threadIdx.x) >> 5;   // 384 waves
  const int m0   = wave * 16;
  const int ln   = lane & 15;
  const int hi   = lane >> 4;
  const int arow = m0 + ln;

  v8f zero8 = {0,0,0,0,0,0,0,0};
  v8f acc[12];
#pragma unroll
  for (int q = 0; q < 12; ++q) acc[q] = zero8;

  const ushort_t* ax = hgb + (size_t)arow * HS + hi * 16;
#pragma unroll
  for (int k0 = 0; k0 < HS; k0 += 32) {
    v16bf af = fragA(ax + k0);
#pragma unroll
    for (int q = 0; q < 12; ++q) {
      int n = q * 16 + ln;
      acc[q] = wmma_bf16(af, fragB(whhb + (size_t)n * HS + k0 + hi * 16), acc[q]);
    }
  }

  const float* gxs = gx + (size_t)sw * (BB * TT) * (3 * HS);
#pragma unroll
  for (int a = 0; a < 4; ++a) {
    int j = a * 16 + ln;
#pragma unroll
    for (int r = 0; r < 8; ++r) {
      int row = m0 + r + hi * 8;
      const float* gi = gxs + (size_t)row * (3 * HS);
      float ghr = acc[a][r]     + bhh[j];
      float ghz = acc[4 + a][r] + bhh[HS + j];
      float ghn = acc[8 + a][r] + bhh[2 * HS + j];
      float rg = sigm(gi[j] + ghr);
      float zg = sigm(gi[HS + j] + ghz);
      float ng = tanhf(gi[2 * HS + j] + rg * ghn);
      float ho = bf2f(hgb[(size_t)row * HS + j]);
      hgb[(size_t)row * HS + j] = f2bfu((1.0f - zg) * ng + zg * ho);
    }
  }
}

// out += h2 @ ls_w^T + ls_b.  A[b][k] = h_gru[(b*24 + k/64)][k%64]. K=1536, N=321.
__global__ void lin_s(const ushort_t* __restrict__ hgb, const ushort_t* __restrict__ wb,
                      const float* __restrict__ bias, float* __restrict__ out) {
  const int lane  = threadIdx.x & 31;
  const int wave  = (blockIdx.x * blockDim.x + threadIdx.x) >> 5;
  const int mtile = wave / 6;
  const int n0    = (wave % 6) * 64;
  const int m0    = mtile * 16;
  const int ln    = lane & 15;
  const int hi    = lane >> 4;
  const int arow  = m0 + ln;

  v8f zero8 = {0,0,0,0,0,0,0,0};
  v8f acc[4];
#pragma unroll
  for (int t = 0; t < 4; ++t) acc[t] = zero8;

  for (int k0 = 0; k0 < TT * HS; k0 += 32) {
    int ka = k0 + hi * 16;                       // 8-elem runs never cross a 64 boundary
    int kb = ka + 8;
    const ushort_t* p0 = hgb + ((size_t)arow * TT + (ka >> 6)) * HS + (ka & 63);
    const ushort_t* p1 = hgb + ((size_t)arow * TT + (kb >> 6)) * HS + (kb & 63);
    v4u a = *(const v4u*)p0;
    v4u b = *(const v4u*)p1;
    v8u ru; ru[0]=a[0]; ru[1]=a[1]; ru[2]=a[2]; ru[3]=a[3];
            ru[4]=b[0]; ru[5]=b[1]; ru[6]=b[2]; ru[7]=b[3];
    v16bf af = __builtin_bit_cast(v16bf, ru);
#pragma unroll
    for (int t = 0; t < 4; ++t) {
      int n  = n0 + t * 16 + ln;
      int nc = n < DD ? n : DD - 1;
      acc[t] = wmma_bf16(af, fragB(wb + (size_t)nc * (TT * HS) + k0 + hi * 16), acc[t]);
    }
  }
#pragma unroll
  for (int t = 0; t < 4; ++t) {
    int n = n0 + t * 16 + ln;
    if (n < DD) {
      float bias_n = bias[n];
#pragma unroll
      for (int r = 0; r < 8; ++r) {
        int m = m0 + r + hi * 8;
        out[m * DD + n] += acc[t][r] + bias_n;
      }
    }
  }
}

// AR branch: out[b][d] += la_b + sum_r x[b, W-24+r, d] * la_w[r]
__global__ void ar_add(const float* __restrict__ x, const float* __restrict__ law,
                       const float* __restrict__ lab, float* __restrict__ out) {
  int idx = blockIdx.x * blockDim.x + threadIdx.x;   // 82176
  int b = idx / DD, d = idx - b * DD;
  float a = lab[0];
#pragma unroll
  for (int r = 0; r < ARN; ++r)
    a += x[((size_t)b * WW + (WW - ARN + r)) * DD + d] * law[r];
  out[idx] += a;
}

extern "C" void kernel_launch(void* const* d_in, const int* in_sizes, int n_in,
                              void* d_out, int out_size, void* d_ws, size_t ws_size,
                              hipStream_t stream) {
  (void)in_sizes; (void)n_in; (void)out_size; (void)ws_size;
  const float* x      = (const float*)d_in[0];
  const float* conv_w = (const float*)d_in[1];
  const float* conv_b = (const float*)d_in[2];
  const float* wih    = (const float*)d_in[3];
  const float* whh    = (const float*)d_in[4];
  const float* bih    = (const float*)d_in[5];
  const float* bhh    = (const float*)d_in[6];
  const float* att_w  = (const float*)d_in[7];
  const float* att_b  = (const float*)d_in[8];
  const float* gwih   = (const float*)d_in[9];
  const float* gwhh   = (const float*)d_in[10];
  const float* gbih   = (const float*)d_in[11];
  const float* gbhh   = (const float*)d_in[12];
  const float* lg_w   = (const float*)d_in[13];
  const float* lg_b   = (const float*)d_in[14];
  const float* ls_w   = (const float*)d_in[15];
  const float* ls_b   = (const float*)d_in[16];
  const float* la_w   = (const float*)d_in[17];
  const float* la_b   = (const float*)d_in[18];
  float* out = (float*)d_out;

  // Workspace layout (float units; bf16 buffers use count/2 floats). ~144 MB.
  float* ws = (float*)d_ws;
  size_t off = 0;
  ushort_t* xb    = (ushort_t*)(ws + off); off += (size_t)BB*(WW+WPAD)*XSTR/2;   // x, padded bf16
  ushort_t* cwb   = (ushort_t*)(ws + off); off += (size_t)HC*KK*XSTR/2;          // conv_w bf16
  ushort_t* gbufb = (ushort_t*)(ws + off); off += (size_t)(WW+GPAD)*BB*HC/2;     // conv out bf16
  ushort_t* wihb  = (ushort_t*)(ws + off); off += (size_t)4*HG*HC/2;
  ushort_t* whhb  = (ushort_t*)(ws + off); off += (size_t)4*HG*HG/2;
  ushort_t* gwihb = (ushort_t*)(ws + off); off += (size_t)3*HS*HC/2;
  ushort_t* gwhhb = (ushort_t*)(ws + off); off += (size_t)3*HS*HS/2;
  ushort_t* lgwb  = (ushort_t*)(ws + off); off += (size_t)DD*HG/2;               // 82176 even
  ushort_t* lswb  = (ushort_t*)(ws + off); off += (size_t)DD*TT*HS/2;            // 493056 even
  ushort_t* hb    = (ushort_t*)(ws + off); off += (size_t)BB*HG/2;               // LSTM h bf16
  ushort_t* attob = (ushort_t*)(ws + off); off += (size_t)BB*HG/2;               // att_out bf16
  ushort_t* hgb   = (ushort_t*)(ws + off); off += (size_t)BB*TT*HS/2;            // GRU h bf16
  float* outs   = ws + off; off += (size_t)WL*BB*HG;                             // LSTM trace f32
  float* Cbuf   = ws + off; off += (size_t)BB*HG;                                // LSTM C f32
  float* gates  = ws + off; off += (size_t)BB*4*HG;                              // gates f32
  float* scores = ws + off; off += (size_t)BB*WL;                                // att logits
  float* gx     = ws + off; off += (size_t)SWN*BB*TT*3*HS;                       // GRU x-gates

  // --- one-time (per launch) packing / zeroing ---
  {
    int ndw;
    ndw = BB*(WW+WPAD)*(XSTR/2);
    pack_x<<<(ndw+255)/256, 256, 0, stream>>>(x, (unsigned*)xb);
    ndw = HC*KK*(XSTR/2);
    pack_cw<<<(ndw+255)/256, 256, 0, stream>>>(conv_w, (unsigned*)cwb);
    ndw = 4*HG*HC/2;
    pack_bf16<<<(ndw+255)/256, 256, 0, stream>>>(wih, (unsigned*)wihb, ndw);
    ndw = 4*HG*HG/2;
    pack_bf16<<<(ndw+255)/256, 256, 0, stream>>>(whh, (unsigned*)whhb, ndw);
    ndw = 3*HS*HC/2;
    pack_bf16<<<(ndw+255)/256, 256, 0, stream>>>(gwih, (unsigned*)gwihb, ndw);
    ndw = 3*HS*HS/2;
    pack_bf16<<<(ndw+255)/256, 256, 0, stream>>>(gwhh, (unsigned*)gwhhb, ndw);
    ndw = DD*HG/2;
    pack_bf16<<<(ndw+255)/256, 256, 0, stream>>>(lg_w, (unsigned*)lgwb, ndw);
    ndw = DD*TT*HS/2;
    pack_bf16<<<(ndw+255)/256, 256, 0, stream>>>(ls_w, (unsigned*)lswb, ndw);
    ndw = GPAD*BB*HC/2;     // zero pad slices of gbufb
    zero_u<<<(ndw+255)/256, 256, 0, stream>>>((unsigned*)gbufb, ndw);
    ndw = BB*HG/2;          // h = 0
    zero_u<<<(ndw+255)/256, 256, 0, stream>>>((unsigned*)hb, ndw);
    ndw = BB*TT*HS/2;       // h_gru = 0
    zero_u<<<(ndw+255)/256, 256, 0, stream>>>((unsigned*)hgb, ndw);
    zero_f<<<(BB*HG+255)/256, 256, 0, stream>>>(Cbuf, BB*HG);
  }

  // Conv branch (implicit GEMM, 5376 waves, 32x64 tiles)
  conv_gemm<<<672, 256, 0, stream>>>(xb, cwb, conv_b, gbufb);

  // LSTM recurrence: per-step gates GEMM (all-bf16 operands) + pointwise update
  for (int t = 0; t < WL; ++t) {
    lstm_gates<<<32, 256, 0, stream>>>(gbufb + (size_t)(GPAD + t)*BB*HC, hb,
                                       wihb, whhb, bih, bhh, gates);
    lstm_point<<<256, 256, 0, stream>>>(gates, hb, Cbuf, outs + (size_t)t*BB*HG);
  }

  // Attention pooling + one more LSTM step
  att_score<<<WL, 256, 0, stream>>>(outs, att_w, att_b, scores);
  att_pool<<<BB, 256, 0, stream>>>(outs, scores, attob);
  lstm_gates<<<32, 256, 0, stream>>>(attob, hb, wihb, whhb, bih, bhh, gates);
  lstm_point<<<256, 256, 0, stream>>>(gates, hb, Cbuf, nullptr);

  // res = h_f @ lg_w^T + lg_b   (writes d_out)
  lin_g<<<12, 256, 0, stream>>>(hb, lgwb, lg_b, out);

  // Skip-GRU branch: bulk x-gates GEMM, then 8 fused recurrent steps
  gru_gx<<<1152, 256, 0, stream>>>(gbufb, gwihb, gbih, gx);
  for (int s = 0; s < SWN; ++s)
    gru_step<<<48, 256, 0, stream>>>(hgb, gwhhb, gbhh, gx, s);
  lin_s<<<12, 256, 0, stream>>>(hgb, lswb, ls_b, out);

  // AR branch
  ar_add<<<321, 256, 0, stream>>>(x, la_w, la_b, out);
}
